// TimeConditionalViTBlock_86947317940586
// MI455X (gfx1250) — matrix operand order repaired and checked by reading, not compiled, not run on hardware
//
#include <hip/hip_runtime.h>

typedef unsigned int   u32;
typedef unsigned short u16;
typedef __attribute__((ext_vector_type(16))) __bf16 v16bf;
typedef __attribute__((ext_vector_type(8)))  float  v8f;
typedef __attribute__((ext_vector_type(4)))  u32    u32x4;
typedef __attribute__((ext_vector_type(2)))  u32    u32x2;
typedef __attribute__((ext_vector_type(4)))  float  f32x4;

#define D_MODEL 1024
#define D_T     256
#define N_HEADS 16
#define D_HEAD  64
#define HIDDEN  4096
#define BATCH   8
#define SEQ     1024
#define NTOK    (BATCH * SEQ)   /* 8192 tokens */

// CDNA5 async global->LDS copies (ASYNCcnt) if this toolchain exposes them.
#if __has_builtin(__builtin_amdgcn_global_load_async_to_lds_b128) && \
    __has_builtin(__builtin_amdgcn_s_wait_asynccnt)
#define HAVE_ASYNC_LDS 1
#else
#define HAVE_ASYNC_LDS 0
#endif

#if HAVE_ASYNC_LDS
#define AS1 __attribute__((address_space(1)))
#define AS3 __attribute__((address_space(3)))
typedef int v4i_vs __attribute__((vector_size(16)));   // matches builtin param type
__device__ __forceinline__ void async_ld_b128(const u16* g, u16* l) {
  __builtin_amdgcn_global_load_async_to_lds_b128(
      (AS1 v4i_vs*)g, (AS3 v4i_vs*)l, 0, 0);
}
__device__ __forceinline__ void async_wait0() {
  __builtin_amdgcn_s_wait_asynccnt(0);
}
#endif

// ---------------------------------------------------------------- helpers

__device__ __forceinline__ u16 f2bf(float f) {
  u32 u = __float_as_uint(f);
  u32 r = u + 0x7FFFu + ((u >> 16) & 1u);   // round-to-nearest-even
  return (u16)(r >> 16);
}
__device__ __forceinline__ float bf2f(u16 x) {
  return __uint_as_float(((u32)x) << 16);
}

union Frag { u32x4 q[2]; v16bf bf; };

// A-matrix fragment, 16x32 bf16 (ISA 7.12.2): lane<16 row=lane, K=0..7,16..23;
// lane>=16 row=lane-16, K=8..15,24..31.  Rows K-contiguous with `stride`.
__device__ __forceinline__ v16bf ld_frag_a(const u16* base, int stride, int koff, int lane) {
  int row = lane & 15;
  int kb  = koff + ((lane >> 4) << 3);
  const u16* p = base + row * stride + kb;
  Frag f;
  f.q[0] = *(const u32x4*)(p);
  f.q[1] = *(const u32x4*)(p + 16);
  return f.bf;
}

// B-matrix fragment, 32x16 bf16 as W[N][K] rows (K contiguous):
// lane<16 col=lane K=0..7,16..23; lane>=16 col=lane-16 K=8..15,24..31.
__device__ __forceinline__ v16bf ld_frag_b(const u16* base, int stride, int koff, int lane) {
  int row = lane & 15;
  int kb  = koff + ((lane >> 4) << 3);
  const u16* p = base + row * stride + kb;
  Frag f;
  f.q[0] = *(const u32x4*)(p);
  f.q[1] = *(const u32x4*)(p + 16);
  return f.bf;
}

__device__ __forceinline__ v8f wmma_bf16(v16bf a, v16bf b, v8f c) {
  return __builtin_amdgcn_wmma_f32_16x16x32_bf16(false, a, false, b, (short)0, c,
                                                 false, false);
}

// ---------------------------------------------------------------- tiny kernels

__global__ __launch_bounds__(256) void f32_to_bf16_kernel(const float* __restrict__ in,
                                                          u16* __restrict__ out, int n4) {
  int i = blockIdx.x * 256 + threadIdx.x;
  if (i >= n4) return;
  f32x4 v = ((const f32x4*)in)[i];
  u32 lo = (u32)f2bf(v[0]) | ((u32)f2bf(v[1]) << 16);
  u32 hi = (u32)f2bf(v[2]) | ((u32)f2bf(v[3]) << 16);
  u32x2 pk = {lo, hi};
  ((u32x2*)out)[i] = pk;
}

// out[b][d] = dot(t[b][:], w[d][:]) + bias[d]
__global__ __launch_bounds__(256) void modulation_kernel(const float* __restrict__ t,
                                                         const float* __restrict__ w,
                                                         const float* __restrict__ bias,
                                                         float* __restrict__ out) {
  int i = blockIdx.x * 256 + threadIdx.x;     // 0 .. 8191
  int b = i >> 10;
  int d = i & (D_MODEL - 1);
  const f32x4* tv = (const f32x4*)(t + (size_t)b * D_T);
  const f32x4* wv = (const f32x4*)(w + (size_t)d * D_T);
  float s = 0.f;
#pragma unroll 8
  for (int j = 0; j < D_T / 4; ++j) {
    f32x4 a = tv[j], c = wv[j];
    s += a[0] * c[0] + a[1] * c[1] + a[2] * c[2] + a[3] * c[3];
  }
  out[i] = s + bias[d];
}

// per-token RMSNorm with time-conditioned gamma/beta; emits bf16 activations
__global__ __launch_bounds__(256) void rmsnorm_mod_kernel(const float* __restrict__ X,
                                                          const float* __restrict__ G,
                                                          const float* __restrict__ Bt,
                                                          u16* __restrict__ out) {
  int m    = blockIdx.x;           // token 0..8191
  int bidx = m >> 10;
  int tid  = threadIdx.x;
  int lane = tid & 31, wid = tid >> 5;
  const float* xr = X + (size_t)m * D_MODEL;
  f32x4 xv = ((const f32x4*)xr)[tid];
  float ss = xv[0] * xv[0] + xv[1] * xv[1] + xv[2] * xv[2] + xv[3] * xv[3];
#pragma unroll
  for (int mk = 1; mk < 32; mk <<= 1) ss += __shfl_xor(ss, mk, 32);
  __shared__ float red[8];
  if (lane == 0) red[wid] = ss;
  __syncthreads();
  float tot = 0.f;
#pragma unroll
  for (int j = 0; j < 8; ++j) tot += red[j];
  float inv = rsqrtf(tot * (1.0f / D_MODEL) + 1e-6f);
  const f32x4 gv = ((const f32x4*)(G  + (size_t)bidx * D_MODEL))[tid];
  const f32x4 bv = ((const f32x4*)(Bt + (size_t)bidx * D_MODEL))[tid];
  float r0 = gv[0] * xv[0] * inv + bv[0];
  float r1 = gv[1] * xv[1] * inv + bv[1];
  float r2 = gv[2] * xv[2] * inv + bv[2];
  float r3 = gv[3] * xv[3] * inv + bv[3];
  u32 lo = (u32)f2bf(r0) | ((u32)f2bf(r1) << 16);
  u32 hi = (u32)f2bf(r2) | ((u32)f2bf(r3) << 16);
  u32x2 pk = {lo, hi};
  ((u32x2*)(out + (size_t)m * D_MODEL))[tid] = pk;
}

// h = silu(g) * u, all bf16; 2 elements / thread
__global__ __launch_bounds__(256) void silu_mul_kernel(const u16* __restrict__ g,
                                                       const u16* __restrict__ u,
                                                       u16* __restrict__ h, int n2) {
  int i = blockIdx.x * 256 + threadIdx.x;
  if (i >= n2) return;
  u32 gv = ((const u32*)g)[i];
  u32 uv = ((const u32*)u)[i];
  float g0 = bf2f((u16)(gv & 0xFFFF)), g1 = bf2f((u16)(gv >> 16));
  float u0 = bf2f((u16)(uv & 0xFFFF)), u1 = bf2f((u16)(uv >> 16));
  float s0 = g0 / (1.0f + __expf(-g0));
  float s1 = g1 / (1.0f + __expf(-g1));
  u32 pk = (u32)f2bf(s0 * u0) | ((u32)f2bf(s1 * u1) << 16);
  ((u32*)h)[i] = pk;
}

// ---------------------------------------------------------------- GEMM

// C[M,N] = A[M,K] @ W[N,K]^T ; A,W bf16 row-major (K contiguous).
// EPI 0: store bf16.  EPI 1: store f32 + residual.
// Block 256 threads = 8 waves -> 128x256 tile; wave = 64x64 (4x4 wmma tiles),
// 16 wmma per 16 ds_load_b128 per K-step.
template <int EPI>
__global__ __launch_bounds__(256) void gemm_bf16_kernel(const u16* __restrict__ A,
                                                        const u16* __restrict__ W,
                                                        const float* __restrict__ res,
                                                        float* __restrict__ outF,
                                                        u16* __restrict__ outB,
                                                        int M, int N, int K) {
  __shared__ u16 As[128 * 40];     // 128 rows x 32 bf16, padded to 40 (80B rows, 16B-aligned)
  __shared__ u16 Ws[256 * 40];     // 256 rows x 32 bf16
  int tid  = threadIdx.x;
  int lane = tid & 31, wid = tid >> 5;
  int wm = (wid >> 2) * 64;        // 0 or 64
  int wn = (wid & 3) * 64;         // 0,64,128,192
  int m0 = blockIdx.y * 128;
  int n0 = blockIdx.x * 256;

  v8f acc[4][4];
#pragma unroll
  for (int mi = 0; mi < 4; ++mi)
#pragma unroll
    for (int ni = 0; ni < 4; ++ni) acc[mi][ni] = {};

  for (int k0 = 0; k0 < K; k0 += 32) {
    __syncthreads();
#if HAVE_ASYNC_LDS
    {
      // A tile: 128 rows x 4 b128 chunks = 512, 2 per thread (async, no VGPR staging)
#pragma unroll
      for (int p = 0; p < 2; ++p) {
        int idx = tid + p * 256;
        int r = idx >> 2, c4 = idx & 3;
        async_ld_b128(A + (size_t)(m0 + r) * K + k0 + c4 * 8, As + r * 40 + c4 * 8);
      }
      // W tile: 256 rows x 4 chunks = 1024, 4 per thread
#pragma unroll
      for (int p = 0; p < 4; ++p) {
        int idx = tid + p * 256;
        int r = idx >> 2, c4 = idx & 3;
        async_ld_b128(W + (size_t)(n0 + r) * K + k0 + c4 * 8, Ws + r * 40 + c4 * 8);
      }
    }
    async_wait0();                 // own wave's async writes landed in LDS
#else
#pragma unroll
    for (int p = 0; p < 2; ++p) {
      int idx = tid + p * 256;
      int r = idx >> 2, c4 = idx & 3;
      *(u32x4*)(As + r * 40 + c4 * 8) =
          *(const u32x4*)(A + (size_t)(m0 + r) * K + k0 + c4 * 8);
    }
#pragma unroll
    for (int p = 0; p < 4; ++p) {
      int idx = tid + p * 256;
      int r = idx >> 2, c4 = idx & 3;
      *(u32x4*)(Ws + r * 40 + c4 * 8) =
          *(const u32x4*)(W + (size_t)(n0 + r) * K + k0 + c4 * 8);
    }
#endif
    if (k0 + 32 < K) {                       // CDNA5 global_prefetch_b8 of next K tile
      __builtin_prefetch(A + (size_t)(m0 + (tid >> 1)) * K + k0 + 32, 0, 1);
      __builtin_prefetch(W + (size_t)(n0 + tid) * K + k0 + 32, 0, 1);
    }
    __syncthreads();               // everyone's tile data visible

    v16bf bfr[4];
#pragma unroll
    for (int ni = 0; ni < 4; ++ni)
      bfr[ni] = ld_frag_b(Ws + (wn + ni * 16) * 40, 40, 0, lane);
#pragma unroll
    for (int mi = 0; mi < 4; ++mi) {
      v16bf af = ld_frag_a(As + (wm + mi * 16) * 40, 40, 0, lane);
#pragma unroll
      for (int ni = 0; ni < 4; ++ni)
        acc[mi][ni] = wmma_bf16(af, bfr[ni], acc[mi][ni]);
    }
  }

  int crow = (lane >> 4) * 8;
  int ccol = lane & 15;
#pragma unroll
  for (int mi = 0; mi < 4; ++mi)
#pragma unroll
    for (int ni = 0; ni < 4; ++ni)
#pragma unroll
      for (int v = 0; v < 8; ++v) {
        int gr = m0 + wm + mi * 16 + v + crow;
        int gc = n0 + wn + ni * 16 + ccol;
        float val = acc[mi][ni][v];
        size_t o = (size_t)gr * N + gc;
        if (EPI == 0) outB[o] = f2bf(val);
        else          outF[o] = val + res[o];
      }
}

// ---------------------------------------------------------------- flash attention

// One block per (b, h, 128-query tile). 8 waves, each owns 16 query rows.
// Online softmax over 32-wide KV tiles; QK^T and PV on WMMA.
__global__ __launch_bounds__(256) void flash_attn_kernel(const u16* __restrict__ Q,
                                                         const u16* __restrict__ Kb,
                                                         const u16* __restrict__ Vb,
                                                         u16* __restrict__ O) {
  __shared__ u16 Ks[32 * 72];       // K tile  [32 kv][64 dh], stride 72 (144B, 16B-aligned)
  __shared__ u16 Vt[64 * 40];       // V^T tile [64 dh][32 kv], stride 40
  __shared__ u16 Ps[8][16 * 40];    // per-wave P staging [16 q][32 kv]

  int b  = blockIdx.z, h = blockIdx.y;
  int q0 = blockIdx.x * 128;
  int tid = threadIdx.x, lane = tid & 31, wid = tid >> 5;
  int qr = q0 + wid * 16;
  size_t tokb = (size_t)b * SEQ;
  int hcol = h * D_HEAD;
  const float scale = 0.125f;       // 1/sqrt(64)

  // Q fragments (A-layout) kept in registers: 2 K-chunks of 32
  v16bf qf[2];
  {
    int qrow = qr + (lane & 15);
    const u16* qp = Q + (tokb + qrow) * D_MODEL + hcol;
    int kb = (lane >> 4) * 8;
#pragma unroll
    for (int c = 0; c < 2; ++c) {
      Frag f;
      f.q[0] = *(const u32x4*)(qp + c * 32 + kb);
      f.q[1] = *(const u32x4*)(qp + c * 32 + kb + 16);
      qf[c] = f.bf;
    }
  }

  float m_run[8], l_run[8];
#pragma unroll
  for (int v = 0; v < 8; ++v) { m_run[v] = -1e30f; l_run[v] = 0.f; }
  v8f accO[4];
#pragma unroll
  for (int t = 0; t < 4; ++t) accO[t] = {};

  for (int kv0 = 0; kv0 < SEQ; kv0 += 32) {
    __syncthreads();
    {   // K tile: 32x64, one b128 per thread
      int r = tid >> 3, c8 = tid & 7;
#if HAVE_ASYNC_LDS
      async_ld_b128(Kb + (tokb + kv0 + r) * D_MODEL + hcol + c8 * 8,
                    Ks + r * 72 + c8 * 8);
#else
      *(u32x4*)(Ks + r * 72 + c8 * 8) =
          *(const u32x4*)(Kb + (tokb + kv0 + r) * D_MODEL + hcol + c8 * 8);
#endif
    }
    for (int i = tid; i < 32 * 64; i += 256) {   // V transposed into LDS
      int r = i >> 6, c = i & 63;
      Vt[c * 40 + r] = Vb[(tokb + kv0 + r) * D_MODEL + hcol + c];
    }
#if HAVE_ASYNC_LDS
    async_wait0();
#endif
    __syncthreads();

    // S = Q @ K^T  (two 16-col tiles, K=64 in 2 chunks)
    v8f s[2];
#pragma unroll
    for (int nt = 0; nt < 2; ++nt) {
      v8f sa = {};
#pragma unroll
      for (int c = 0; c < 2; ++c) {
        v16bf kf = ld_frag_b(Ks + nt * 16 * 72, 72, c * 32, lane);
        sa = wmma_bf16(qf[c], kf, sa);
      }
      s[nt] = sa;
    }

    // online softmax (row stats across 16-lane halves)
    float alpha[8];
#pragma unroll
    for (int v = 0; v < 8; ++v) {
      float s0 = s[0][v] * scale, s1 = s[1][v] * scale;
      float mx = fmaxf(s0, s1);
#pragma unroll
      for (int mk = 1; mk < 16; mk <<= 1) mx = fmaxf(mx, __shfl_xor(mx, mk, 32));
      float mnew = fmaxf(m_run[v], mx);
      float a = __expf(m_run[v] - mnew);
      float p0 = __expf(s0 - mnew), p1 = __expf(s1 - mnew);
      s[0][v] = p0; s[1][v] = p1;
      float ps = p0 + p1;
#pragma unroll
      for (int mk = 1; mk < 16; mk <<= 1) ps += __shfl_xor(ps, mk, 32);
      l_run[v] = l_run[v] * a + ps;
      m_run[v] = mnew;
      alpha[v] = a;
    }

    // C-layout -> A-layout via per-wave LDS (same-wave DS ops are in-order)
    u16* pp = Ps[wid];
    {
      int crow = (lane >> 4) * 8, ccol = lane & 15;
#pragma unroll
      for (int v = 0; v < 8; ++v) {
        pp[(v + crow) * 40 + ccol]      = f2bf(s[0][v]);
        pp[(v + crow) * 40 + 16 + ccol] = f2bf(s[1][v]);
      }
    }
    __builtin_amdgcn_wave_barrier();
    v16bf pf = ld_frag_a(pp, 40, 0, lane);

    // O = O*alpha + P @ V   (4 dh tiles of 16)
#pragma unroll
    for (int t = 0; t < 4; ++t) {
#pragma unroll
      for (int v = 0; v < 8; ++v) accO[t][v] *= alpha[v];
      v16bf vf = ld_frag_b(Vt + t * 16 * 40, 40, 0, lane);
      accO[t] = wmma_bf16(pf, vf, accO[t]);
    }
  }

  // normalize + store bf16
  int crow = (lane >> 4) * 8, ccol = lane & 15;
#pragma unroll
  for (int t = 0; t < 4; ++t)
#pragma unroll
    for (int v = 0; v < 8; ++v) {
      float val = accO[t][v] / l_run[v];
      int row = qr + v + crow;
      int col = hcol + t * 16 + ccol;
      O[(tokb + row) * D_MODEL + col] = f2bf(val);
    }
}

// ---------------------------------------------------------------- launcher

extern "C" void kernel_launch(void* const* d_in, const int* in_sizes, int n_in,
                              void* d_out, int out_size, void* d_ws, size_t ws_size,
                              hipStream_t stream) {
  (void)in_sizes; (void)n_in; (void)out_size; (void)ws_size;
  const float* x      = (const float*)d_in[0];
  const float* t      = (const float*)d_in[1];
  const float* Wq     = (const float*)d_in[2];
  const float* Wk     = (const float*)d_in[3];
  const float* Wv     = (const float*)d_in[4];
  const float* Wo     = (const float*)d_in[5];
  const float* g1w    = (const float*)d_in[6];
  const float* g1b    = (const float*)d_in[7];
  const float* b1w    = (const float*)d_in[8];
  const float* b1b    = (const float*)d_in[9];
  const float* g2w    = (const float*)d_in[10];
  const float* g2b    = (const float*)d_in[11];
  const float* b2w    = (const float*)d_in[12];
  const float* b2b    = (const float*)d_in[13];
  const float* gate_w = (const float*)d_in[14];
  const float* hid_w  = (const float*)d_in[15];
  const float* out_w  = (const float*)d_in[16];
  float* out = (float*)d_out;

  char* base = (char*)d_ws;
  size_t off = 0;
  auto alloc = [&](size_t bytes) -> char* {
    char* p = base + off;
    off += (bytes + 255) & ~(size_t)255;
    return p;
  };
  const size_t DD = (size_t)D_MODEL * D_MODEL;     // 1M
  const size_t DH = (size_t)HIDDEN * D_MODEL;      // 4M
  u16* wq_b  = (u16*)alloc(DD * 2);
  u16* wk_b  = (u16*)alloc(DD * 2);
  u16* wv_b  = (u16*)alloc(DD * 2);
  u16* wo_b  = (u16*)alloc(DD * 2);
  u16* gw_b  = (u16*)alloc(DH * 2);
  u16* hw_b  = (u16*)alloc(DH * 2);
  u16* ow_b  = (u16*)alloc(DH * 2);
  float* g1  = (float*)alloc((size_t)BATCH * D_MODEL * 4);
  float* b1  = (float*)alloc((size_t)BATCH * D_MODEL * 4);
  float* g2  = (float*)alloc((size_t)BATCH * D_MODEL * 4);
  float* b2  = (float*)alloc((size_t)BATCH * D_MODEL * 4);
  u16* xn    = (u16*)alloc((size_t)NTOK * D_MODEL * 2);
  u16* qb    = (u16*)alloc((size_t)NTOK * D_MODEL * 2);
  u16* kb    = (u16*)alloc((size_t)NTOK * D_MODEL * 2);
  u16* vb    = (u16*)alloc((size_t)NTOK * D_MODEL * 2);
  u16* ob    = (u16*)alloc((size_t)NTOK * D_MODEL * 2);
  float* x1  = (float*)alloc((size_t)NTOK * D_MODEL * 4);
  u16* gbuf  = (u16*)alloc((size_t)NTOK * HIDDEN * 2);
  u16* ubuf  = (u16*)alloc((size_t)NTOK * HIDDEN * 2);

  // 1) weights -> bf16
  auto conv = [&](const float* src, u16* dst, size_t n) {
    int n4 = (int)(n / 4);
    f32_to_bf16_kernel<<<n4 / 256, 256, 0, stream>>>(src, dst, n4);
  };
  conv(Wq, wq_b, DD);  conv(Wk, wk_b, DD);  conv(Wv, wv_b, DD);  conv(Wo, wo_b, DD);
  conv(gate_w, gw_b, DH);  conv(hid_w, hw_b, DH);  conv(out_w, ow_b, DH);

  // 2) time-conditioned gamma/beta
  const int modBlocks = (BATCH * D_MODEL) / 256;
  modulation_kernel<<<modBlocks, 256, 0, stream>>>(t, g1w, g1b, g1);
  modulation_kernel<<<modBlocks, 256, 0, stream>>>(t, b1w, b1b, b1);
  modulation_kernel<<<modBlocks, 256, 0, stream>>>(t, g2w, g2b, g2);
  modulation_kernel<<<modBlocks, 256, 0, stream>>>(t, b2w, b2b, b2);

  // 3) norm1 -> bf16
  rmsnorm_mod_kernel<<<NTOK, 256, 0, stream>>>(x, g1, b1, xn);

  // 4) QKV projections (bf16 WMMA)
  dim3 gQKV(D_MODEL / 256, NTOK / 128);
  gemm_bf16_kernel<0><<<gQKV, 256, 0, stream>>>(xn, wq_b, nullptr, nullptr, qb,
                                                NTOK, D_MODEL, D_MODEL);
  gemm_bf16_kernel<0><<<gQKV, 256, 0, stream>>>(xn, wk_b, nullptr, nullptr, kb,
                                                NTOK, D_MODEL, D_MODEL);
  gemm_bf16_kernel<0><<<gQKV, 256, 0, stream>>>(xn, wv_b, nullptr, nullptr, vb,
                                                NTOK, D_MODEL, D_MODEL);

  // 5) flash attention
  dim3 gAtt(SEQ / 128, N_HEADS, BATCH);
  flash_attn_kernel<<<gAtt, 256, 0, stream>>>(qb, kb, vb, ob);

  // 6) output projection + residual -> x1 (f32)
  gemm_bf16_kernel<1><<<gQKV, 256, 0, stream>>>(ob, wo_b, x, x1, nullptr,
                                                NTOK, D_MODEL, D_MODEL);

  // 7) norm2 -> bf16
  rmsnorm_mod_kernel<<<NTOK, 256, 0, stream>>>(x1, g2, b2, xn);

  // 8) MLP up projections
  dim3 gUp(HIDDEN / 256, NTOK / 128);
  gemm_bf16_kernel<0><<<gUp, 256, 0, stream>>>(xn, gw_b, nullptr, nullptr, gbuf,
                                               NTOK, HIDDEN, D_MODEL);
  gemm_bf16_kernel<0><<<gUp, 256, 0, stream>>>(xn, hw_b, nullptr, nullptr, ubuf,
                                               NTOK, HIDDEN, D_MODEL);

  // 9) SwiGLU elementwise (h overwrites gbuf)
  {
    int n2 = (int)((size_t)NTOK * HIDDEN / 2);
    silu_mul_kernel<<<n2 / 256, 256, 0, stream>>>(gbuf, ubuf, gbuf, n2);
  }

  // 10) down projection + residual -> out (f32)
  dim3 gDown(D_MODEL / 256, NTOK / 128);
  gemm_bf16_kernel<1><<<gDown, 256, 0, stream>>>(gbuf, ow_b, x1, out, nullptr,
                                                 NTOK, D_MODEL, HIDDEN);
}
